// PartitionAttentionLayer_11244224381303
// MI455X (gfx1250) — matrix operand order repaired and checked by reading, not compiled, not run on hardware
//
#include <hip/hip_runtime.h>
#include <hip/hip_bf16.h>
#include <math.h>

typedef _Float16 half_t;
typedef __attribute__((ext_vector_type(16))) _Float16 v16h;
typedef __attribute__((ext_vector_type(8)))  float    v8f;

#define SEQn   64
#define CDIM   256
#define NHEADS 8
#define DH     32
#define SCALEK 0.0625f   // C**-0.5 = 1/16

// ---------------- workspace layout (bytes) ----------------
#define WS_QKVT   0            // [768][256]  f16  (qkv_w transposed)
#define WS_MERGET 393216       // [256][256]  f16
#define WS_W1T    524288       // [1024][256] f16
#define WS_W2T    1048576      // [256][1024] f16
#define WS_RELB   1572864      // [8][64][64] f32 rel-pos bias per head

// ---------------- LDS layout (bytes) ----------------
#define LDS_X    0        // [64][256] f32 residual / running activation
#define LDS_H    65536    // [64][256] f16 LN out;   reused: VT [8][32][64] f16
#define LDS_Q    98304    // [64][256] f16 Q;        reused: ATT [8][64][64] f16 (spans Q+K)
#define LDS_K    131072   // [64][256] f16 K(scaled)
#define LDS_V    163840   // [64][256] f16 V;        reused: AO [64][256] f16
#define LDS_S    196608   // [8][64][64] f32 scores; reused: HID [64][1024] f16
#define LDS_TOTAL 327680  // 320 KB == WGP LDS

// A fragment, 16(M) x 32(K), row-major contiguous-K storage.
// ISA A layout: lane 0-15 row M=lane holds K={kb..kb+7, 16+kb..16+kb+7}, kb=8*(lane>>4).
__device__ __forceinline__ v16h load_fragA(const half_t* __restrict__ base, int ld) {
  int lane = threadIdx.x & 31;
  int row  = lane & 15;
  int kb   = (lane >> 4) << 3;
  const half_t* p = base + row * ld + kb;
  union { v16h h; uint4 q[2]; } f;
  f.q[0] = *(const uint4*)(p);
  f.q[1] = *(const uint4*)(p + 16);
  return f.h;
}

// B fragment, 32(K) x 16(N), sourced from [N][K]-stored operand (contiguous K).
// B layout matches A: lane 0-15 col N=lane holds K={kb..kb+7, 16+kb..16+kb+7},
// kb=8*(lane>>4).
__device__ __forceinline__ v16h load_fragB(const half_t* __restrict__ base, int ld) {
  int lane = threadIdx.x & 31;
  int col  = lane & 15;
  int kb   = (lane >> 4) << 3;
  const half_t* p = base + col * ld + kb;
  union { v16h h; uint4 q[2]; } f;
  f.q[0] = *(const uint4*)(p);
  f.q[1] = *(const uint4*)(p + 16);
  return f.h;
}

__device__ __forceinline__ v8f wmma_f16(v16h a, v16h b, v8f c) {
  return __builtin_amdgcn_wmma_f32_16x16x32_f16(false, a, false, b, (short)0, c, false, false);
}

#define ZACC {0.f,0.f,0.f,0.f,0.f,0.f,0.f,0.f}

// =========== prep: transpose weights to fp16 [N][K], build rel bias ===========
__global__ __launch_bounds__(256) void prep_kernel(
    const float* __restrict__ qkv_w, const float* __restrict__ merge_w,
    const float* __restrict__ w1,    const float* __restrict__ w2,
    const float* __restrict__ bias_table, char* __restrict__ ws) {
  int idx = blockIdx.x * 256 + threadIdx.x;
  half_t* qkvT = (half_t*)(ws + WS_QKVT);
  half_t* mT   = (half_t*)(ws + WS_MERGET);
  half_t* w1T  = (half_t*)(ws + WS_W1T);
  half_t* w2T  = (half_t*)(ws + WS_W2T);
  float*  relb = (float*) (ws + WS_RELB);
  if (idx < 768*256) { int n = idx >> 8, k = idx & 255; qkvT[n*256+k] = (half_t)qkv_w[k*768+n]; return; }
  idx -= 768*256;
  if (idx < 256*256) { int n = idx >> 8, k = idx & 255; mT[n*256+k] = (half_t)merge_w[k*256+n]; return; }
  idx -= 256*256;
  if (idx < 1024*256){ int n = idx >> 8, k = idx & 255; w1T[n*256+k] = (half_t)w1[k*1024+n]; return; }
  idx -= 1024*256;
  if (idx < 256*1024){ int n = idx >> 10, k = idx & 1023; w2T[n*1024+k] = (half_t)w2[k*256+n]; return; }
  idx -= 256*1024;
  if (idx < NHEADS*64*64) {
    int h = idx >> 12, rem = idx & 4095, i = rem >> 6, j = rem & 63;
    int ry = (i >> 3) - (j >> 3) + 7;
    int rx = (i & 7)  - (j & 7)  + 7;
    relb[idx] = bias_table[(ry * 15 + rx) * NHEADS + h];
  }
}

// =========== fused per-window transformer block ===========
__global__ __launch_bounds__(256) void swin_block_kernel(
    const float* __restrict__ x,
    const float* __restrict__ ln1_g, const float* __restrict__ ln1_b,
    const float* __restrict__ qkv_b, const float* __restrict__ merge_b,
    const float* __restrict__ ln2_g, const float* __restrict__ ln2_b,
    const float* __restrict__ mlp_b1, const float* __restrict__ mlp_b2,
    const char* __restrict__ ws, float* __restrict__ out) {
  __shared__ __align__(16) char smem[LDS_TOTAL];
  float*  X   = (float*)(smem + LDS_X);
  half_t* H   = (half_t*)(smem + LDS_H);
  half_t* Qb  = (half_t*)(smem + LDS_Q);
  half_t* Kb  = (half_t*)(smem + LDS_K);
  half_t* Vb  = (half_t*)(smem + LDS_V);
  float*  S   = (float*)(smem + LDS_S);
  half_t* ATT = (half_t*)(smem + LDS_Q);   // overlays Q+K after scores consumed
  half_t* VT  = (half_t*)(smem + LDS_H);   // overlays H after QKV GEMM
  half_t* AO  = (half_t*)(smem + LDS_V);   // overlays V after attn-out
  half_t* HID = (half_t*)(smem + LDS_S);   // overlays scores for MLP

  const half_t* qkvT = (const half_t*)(ws + WS_QKVT);
  const half_t* mT   = (const half_t*)(ws + WS_MERGET);
  const half_t* w1T  = (const half_t*)(ws + WS_W1T);
  const half_t* w2T  = (const half_t*)(ws + WS_W2T);
  const float*  relb = (const float*)(ws + WS_RELB);

  const int t = threadIdx.x, lane = t & 31, wave = t >> 5;
  const int blk = blockIdx.x;
  const int b = blk >> 6, g = blk & 63, gy = g >> 3, gx = g & 7;
  const size_t base = (size_t)b * (CDIM * 4096) + (size_t)gy * 512 + (size_t)gx * 8;
  const float* xwin = x + base;

  const int n = lane & 15, mb = (lane >> 4) * 8;   // C/D fragment element owner

  // ---- Phase 1: gather window [64 tok][256 ch] into LDS (b128 row segments) ----
  for (int it = 0; it < 16; ++it) {
    int q = it * 256 + t;            // 4096 float4 quads: c*16 + py*2 + half
    int c = q >> 4, rem = q & 15, py = rem >> 1, hf = rem & 1;
    float4 v4 = *(const float4*)(xwin + (size_t)c * 4096 + py * 64 + hf * 4);
    int tok0 = py * 8 + hf * 4;
    X[(tok0 + 0) * CDIM + c] = v4.x;
    X[(tok0 + 1) * CDIM + c] = v4.y;
    X[(tok0 + 2) * CDIM + c] = v4.z;
    X[(tok0 + 3) * CDIM + c] = v4.w;
  }
  __syncthreads();

  // ---- Phase 2: LayerNorm 1 -> H (fp16) ----
  if (t < SEQn) {
    float mu = 0.f;
    for (int c = 0; c < CDIM; ++c) mu += X[t * CDIM + c];
    mu *= (1.f / CDIM);
    float var = 0.f;
    for (int c = 0; c < CDIM; ++c) { float d = X[t * CDIM + c] - mu; var += d * d; }
    float rs = rsqrtf(var * (1.f / CDIM) + 1e-5f);
    for (int c = 0; c < CDIM; ++c)
      H[t * CDIM + c] = (half_t)((X[t * CDIM + c] - mu) * rs * ln1_g[c] + ln1_b[c]);
  }
  __syncthreads();

  // ---- Phase 3: QKV GEMM [64,256]x[256,768] ----
  // Each wave owns N tiles; B fragments (global/L2) cached in registers and
  // reused across all 4 M tiles -> every weight byte read once per workgroup.
  for (int nt = wave; nt < 48; nt += 8) {
    v16h bF[8];
#pragma unroll
    for (int ks = 0; ks < 8; ++ks)
      bF[ks] = load_fragB(qkvT + nt * 16 * CDIM + ks * 32, CDIM);
    int col = nt * 16 + n;
    int region = col >> 8;                       // 0=Q 1=K 2=V (branch-free)
    half_t* dst = (half_t*)(smem + LDS_Q + (region << 15));
    int cc = col & 255;
    float bb = qkv_b[col];
    float sc = (region == 1) ? SCALEK : 1.f;
#pragma unroll
    for (int mt = 0; mt < 4; ++mt) {
      v8f acc = ZACC;
#pragma unroll
      for (int ks = 0; ks < 8; ++ks)
        acc = wmma_f16(load_fragA(H + mt * 16 * CDIM + ks * 32, CDIM), bF[ks], acc);
#pragma unroll
      for (int r = 0; r < 8; ++r)
        dst[(mt * 16 + mb + r) * CDIM + cc] = (half_t)((acc[r] + bb) * sc);
    }
  }
  __syncthreads();

  // ---- Phase 4: transpose V -> VT [head][d][tok] (H region is dead) ----
  for (int it = 0; it < 64; ++it) {
    int idx = it * 256 + t;             // head*2048 + d*64 + j
    int h = idx >> 11, rem = idx & 2047, d = rem >> 6, j = rem & 63;
    VT[idx] = Vb[j * CDIM + h * DH + d];
  }
  __syncthreads();

  // ---- Phase 5: scores = q @ k^T (+rel bias) -> S fp32 ----
  for (int job = wave; job < NHEADS * 16; job += 8) {
    int h = job >> 4, mt = (job >> 2) & 3, nt = job & 3;
    v16h a  = load_fragA(Qb + mt * 16 * CDIM + h * DH, CDIM);
    v16h bw = load_fragB(Kb + nt * 16 * CDIM + h * DH, CDIM);
    v8f acc = ZACC;
    acc = wmma_f16(a, bw, acc);
#pragma unroll
    for (int r = 0; r < 8; ++r) {
      int i = mt * 16 + mb + r, j = nt * 16 + n;
      S[h * 4096 + i * 64 + j] = acc[r] + relb[h * 4096 + i * 64 + j];
    }
  }
  __syncthreads();

  // ---- Phase 6: row softmax; write attn fp16 over dead Q/K region ----
#pragma unroll
  for (int rr = 0; rr < 2; ++rr) {
    int row = t * 2 + rr;               // (head*64 + i), 512 rows total
    const float* sp = S + row * 64;
    float mx = -3.0e38f;
    for (int j = 0; j < 64; ++j) mx = fmaxf(mx, sp[j]);
    float sum = 0.f;
    for (int j = 0; j < 64; ++j) sum += __expf(sp[j] - mx);
    float inv = 1.f / sum;
    half_t* ap = ATT + row * 64;
    for (int j = 0; j < 64; ++j) ap[j] = (half_t)(__expf(sp[j] - mx) * inv);
  }
  __syncthreads();

  // ---- Phase 7: out = attn @ v  (A=ATT [64x64], B=VT [d][tok]) -> AO ----
  for (int job = wave; job < NHEADS * 8; job += 8) {
    int h = job >> 3, mt = (job >> 1) & 3, nt = job & 1;
    v8f acc = ZACC;
#pragma unroll
    for (int ks = 0; ks < 2; ++ks) {
      v16h a  = load_fragA(ATT + h * 4096 + mt * 16 * 64 + ks * 32, 64);
      v16h bw = load_fragB(VT + h * 2048 + nt * 16 * 64 + ks * 32, 64);
      acc = wmma_f16(a, bw, acc);
    }
#pragma unroll
    for (int r = 0; r < 8; ++r)
      AO[(mt * 16 + mb + r) * CDIM + h * DH + nt * 16 + n] = (half_t)acc[r];
  }
  __syncthreads();

  // ---- Phase 8: merge GEMM + residual into X (B cached per N tile) ----
  for (int nt = wave; nt < 16; nt += 8) {
    v16h bF[8];
#pragma unroll
    for (int ks = 0; ks < 8; ++ks)
      bF[ks] = load_fragB(mT + nt * 16 * CDIM + ks * 32, CDIM);
    float bb = merge_b[nt * 16 + n];
#pragma unroll
    for (int mt = 0; mt < 4; ++mt) {
      v8f acc = ZACC;
#pragma unroll
      for (int ks = 0; ks < 8; ++ks)
        acc = wmma_f16(load_fragA(AO + mt * 16 * CDIM + ks * 32, CDIM), bF[ks], acc);
#pragma unroll
      for (int r = 0; r < 8; ++r) {
        int i = mt * 16 + mb + r, c = nt * 16 + n;
        X[i * CDIM + c] += acc[r] + bb;
      }
    }
  }
  __syncthreads();

  // ---- Phase 9: LayerNorm 2 -> H ----
  if (t < SEQn) {
    float mu = 0.f;
    for (int c = 0; c < CDIM; ++c) mu += X[t * CDIM + c];
    mu *= (1.f / CDIM);
    float var = 0.f;
    for (int c = 0; c < CDIM; ++c) { float d = X[t * CDIM + c] - mu; var += d * d; }
    float rs = rsqrtf(var * (1.f / CDIM) + 1e-5f);
    for (int c = 0; c < CDIM; ++c)
      H[t * CDIM + c] = (half_t)((X[t * CDIM + c] - mu) * rs * ln2_g[c] + ln2_b[c]);
  }
  __syncthreads();

  // ---- Phase 10: MLP1 [64,256]x[256,1024] + exact GELU -> HID ----
  for (int nt = wave; nt < 64; nt += 8) {
    v16h bF[8];
#pragma unroll
    for (int ks = 0; ks < 8; ++ks)
      bF[ks] = load_fragB(w1T + nt * 16 * CDIM + ks * 32, CDIM);
    float bb = mlp_b1[nt * 16 + n];
#pragma unroll
    for (int mt = 0; mt < 4; ++mt) {
      v8f acc = ZACC;
#pragma unroll
      for (int ks = 0; ks < 8; ++ks)
        acc = wmma_f16(load_fragA(H + mt * 16 * CDIM + ks * 32, CDIM), bF[ks], acc);
#pragma unroll
      for (int r = 0; r < 8; ++r) {
        int i = mt * 16 + mb + r;
        float v = acc[r] + bb;
        float gelu = 0.5f * v * (1.f + erff(v * 0.70710678118f));
        HID[i * 1024 + nt * 16 + n] = (half_t)gelu;
      }
    }
  }
  __syncthreads();

  // ---- Phase 11: MLP2 [64,1024]x[1024,256] + residual -> departition store ----
  // K=1024: B cache chunked into 4 K-groups; 4 persistent M accumulators.
  {
    float* ybase = out + base;
    for (int nt = wave; nt < 16; nt += 8) {
      v8f acc[4] = {ZACC, ZACC, ZACC, ZACC};
      for (int kc = 0; kc < 4; ++kc) {
        v16h bF[8];
#pragma unroll
        for (int k8 = 0; k8 < 8; ++k8)
          bF[k8] = load_fragB(w2T + nt * 16 * 1024 + (kc * 8 + k8) * 32, 1024);
#pragma unroll
        for (int mt = 0; mt < 4; ++mt)
#pragma unroll
          for (int k8 = 0; k8 < 8; ++k8)
            acc[mt] = wmma_f16(load_fragA(HID + mt * 16 * 1024 + (kc * 8 + k8) * 32, 1024),
                               bF[k8], acc[mt]);
      }
      float bb = mlp_b2[nt * 16 + n];
#pragma unroll
      for (int mt = 0; mt < 4; ++mt) {
#pragma unroll
        for (int r = 0; r < 8; ++r) {
          int i = mt * 16 + mb + r, c = nt * 16 + n;
          float val = acc[mt][r] + bb + X[i * CDIM + c];
          ybase[(size_t)c * 4096 + (i >> 3) * 64 + (i & 7)] = val;
        }
      }
    }
  }
}

extern "C" void kernel_launch(void* const* d_in, const int* in_sizes, int n_in,
                              void* d_out, int out_size, void* d_ws, size_t ws_size,
                              hipStream_t stream) {
  const float* x          = (const float*)d_in[0];
  const float* ln1_g      = (const float*)d_in[1];
  const float* ln1_b      = (const float*)d_in[2];
  const float* qkv_w      = (const float*)d_in[3];
  const float* qkv_b      = (const float*)d_in[4];
  const float* merge_w    = (const float*)d_in[5];
  const float* merge_b    = (const float*)d_in[6];
  const float* bias_table = (const float*)d_in[7];
  const float* ln2_g      = (const float*)d_in[8];
  const float* ln2_b      = (const float*)d_in[9];
  const float* mlp_w1     = (const float*)d_in[10];
  const float* mlp_b1     = (const float*)d_in[11];
  const float* mlp_w2     = (const float*)d_in[12];
  const float* mlp_b2     = (const float*)d_in[13];
  char* ws = (char*)d_ws;
  float* out = (float*)d_out;

  // total prep elements: 196608+65536+262144+262144+32768 = 819200 -> 3200 blocks
  prep_kernel<<<3200, 256, 0, stream>>>(qkv_w, merge_w, mlp_w1, mlp_w2, bias_table, ws);
  swin_block_kernel<<<2048, 256, 0, stream>>>(x, ln1_g, ln1_b, qkv_b, merge_b,
                                              ln2_g, ln2_b, mlp_b1, mlp_b2, ws, out);
}
